// Heat2D_75453985457178
// MI455X (gfx1250) — matrix operand order repaired and checked by reading, not compiled, not run on hardware
//
#include <hip/hip_runtime.h>

// ---------------------------------------------------------------------------
// Heat2D spectral block for MI455X (gfx1250), wave32 + WMMA bf16 16x16x32.
// B=C=H=W=KD=64.  All heavy contractions are 64x64x64 GEMMs done with
// v_wmma_f32_16x16x32_bf16 (8 waves/block, 2 tiles/wave, K-loop of 2x32).
// Intermediates live in one 32MB bf16 buffer that stays hot in the 192MB L2.
// ---------------------------------------------------------------------------

#define LDB 72   // LDS row stride (bf16 elems) for staged A / B^T tiles (16B-aligned rows)

typedef __bf16 bf16;
typedef __attribute__((ext_vector_type(8)))  bf16  v8bf;
typedef __attribute__((ext_vector_type(16))) bf16  v16bf;
typedef __attribute__((ext_vector_type(8)))  float v8f;

__device__ __forceinline__ bf16 f2bf(float f) {
  unsigned u = __builtin_bit_cast(unsigned, f);
  unsigned r = u + 0x7FFFu + ((u >> 16) & 1u);   // round-to-nearest-even
  return __builtin_bit_cast(bf16, (unsigned short)(r >> 16));
}

__device__ __forceinline__ void zero_acc(v8f acc[2]) {
#pragma unroll
  for (int t = 0; t < 2; ++t)
#pragma unroll
    for (int e = 0; e < 8; ++e) acc[t][e] = 0.0f;
}

// C(64x64 f32) = A(64x64) * B(64x64).
// lA : A row-major [m][k], stride LDB.  lBT : B transposed [n][k], stride LDB.
// 8 waves: wave w computes tiles (mi = w>>1, ni = (w&1)*2 + {0,1}).
// Fragments are two ds_load_b128 per lane (A and B: K 0..7 / 16..23 split by lane hi),
// matching the CDNA5 16-bit WMMA VGPR layouts.
__device__ __forceinline__ void wmma64(const bf16* lA, const bf16* lBT,
                                       v8f acc[2], int wave, int lane) {
  const int mi = wave >> 1;
  const int nb = (wave & 1) << 1;
  const int lr = lane & 15;
  const int hi = lane >> 4;
#pragma unroll
  for (int ks = 0; ks < 64; ks += 32) {
    const bf16* pa = lA + (mi * 16 + lr) * LDB + ks + (hi << 3);
    v8bf alo = *(const v8bf*)pa;
    v8bf ahi = *(const v8bf*)(pa + 16);
    v16bf a;
#pragma unroll
    for (int e = 0; e < 8; ++e) { a[e] = alo[e]; a[e + 8] = ahi[e]; }
#pragma unroll
    for (int t = 0; t < 2; ++t) {
      const bf16* pb = lBT + ((nb + t) * 16 + lr) * LDB + ks + (hi << 3);
      v8bf blo = *(const v8bf*)pb;
      v8bf bhi = *(const v8bf*)(pb + 16);
      v16bf bm;
#pragma unroll
      for (int e = 0; e < 8; ++e) { bm[e] = blo[e]; bm[e + 8] = bhi[e]; }
      acc[t] = __builtin_amdgcn_wmma_f32_16x16x32_bf16(
          false, a, false, bm, (short)0, acc[t], false, false);
    }
  }
}

// Scatter accumulators to global bf16 tile: element (m,n) -> base[m*rstride + n].
// D layout: VGPR r -> m = r + 8*(lane>>4), n = lane&15 (per 16x16 tile).
__device__ __forceinline__ void store_acc_bf16(const v8f acc[2], int wave, int lane,
                                               bf16* base, int rstride,
                                               const float* bias) {
  const int m0 = (wave >> 1) * 16 + ((lane >> 4) << 3);
  const int nb = (wave & 1) << 1;
#pragma unroll
  for (int t = 0; t < 2; ++t) {
    const int n = (nb + t) * 16 + (lane & 15);
    const float bv = bias ? bias[n] : 0.0f;
#pragma unroll
    for (int r = 0; r < 8; ++r)
      base[(size_t)(m0 + r) * rstride + n] = f2bf(acc[t][r] + bv);
  }
}

// ---------------------------------------------------------------------------
// K0a: small tables -> bf16: cos matrix (symmetric), Wk^T, Wo^T.
// ---------------------------------------------------------------------------
__global__ void k_tables(const float* __restrict__ Wk, const float* __restrict__ Wo,
                         bf16* __restrict__ cosb, bf16* __restrict__ WkT,
                         bf16* __restrict__ WoT) {
  const int i = blockIdx.x, j = threadIdx.x;
  const float PI = 3.14159265358979323846f;
  cosb[i * 64 + j] = f2bf(__cosf((float)(i * j) * (PI / 64.0f)) * (1.0f / 64.0f));
  WkT[i * 64 + j]  = f2bf(Wk[j * 64 + i]);  // WkT[kd][c]  = Wk[c][kd]
  WoT[i * 64 + j]  = f2bf(Wo[j * 64 + i]);  // WoT[c][kd]  = Wo[kd][c]
}

// ---------------------------------------------------------------------------
// K0b: z = LN(freq)@Wz + bz ;  wexp = decay^z = exp(z*(-(h+w)/64 - ln64)) ;
//      sz = silu(z).  One block per (h,w), 64 threads (c).
// ---------------------------------------------------------------------------
__global__ void k_zgate(const float* __restrict__ freq,
                        const float* __restrict__ g_z, const float* __restrict__ b_z,
                        const float* __restrict__ Wz, const float* __restrict__ bz,
                        float* __restrict__ wexp, float* __restrict__ sz) {
  __shared__ float s0[64], s1[64], snrm[64];
  const int hw = blockIdx.x;
  const int h = hw >> 6, w = hw & 63;
  const int c = threadIdx.x;
  const float f = freq[(size_t)hw * 64 + c];
  s0[c] = f; s1[c] = f * f;
  __syncthreads();
  for (int s = 32; s > 0; s >>= 1) {
    if (c < s) { s0[c] += s0[c + s]; s1[c] += s1[c + s]; }
    __syncthreads();
  }
  const float mu = s0[0] * (1.0f / 64.0f);
  const float var = s1[0] * (1.0f / 64.0f) - mu * mu;
  const float rstd = rsqrtf(var + 1e-5f);
  snrm[c] = (f - mu) * rstd * g_z[c] + b_z[c];
  __syncthreads();
  float zv = bz[c];
  for (int k = 0; k < 64; ++k) zv += snrm[k] * Wz[k * 64 + c];
  const float lg = -((float)(h + w)) * (1.0f / 64.0f) - 4.1588830833596715f; // ln(64)
  wexp[(size_t)hw * 64 + c] = __expf(zv * lg);
  sz[(size_t)hw * 64 + c]   = zv / (1.0f + __expf(-zv));
}

// ---------------------------------------------------------------------------
// K1: per (b,h): transpose x[b,:,h,:] -> [w][c], +freq, LN over c, GEMM @ Wk.
// Writes k (bf16) to buf[b][h][w][kd].
// ---------------------------------------------------------------------------
__global__ void k_input(const float* __restrict__ x, const float* __restrict__ freq,
                        const float* __restrict__ g_k, const float* __restrict__ b_k,
                        const float* __restrict__ bk, const bf16* __restrict__ WkT,
                        bf16* __restrict__ buf) {
  __shared__ float xin[64 * 65];
  __shared__ __align__(16) bf16 lA[64 * LDB];
  __shared__ __align__(16) bf16 lB[64 * LDB];
  __shared__ float mu[64], rstd[64];
  const int tid = threadIdx.x;
  const int b = blockIdx.x >> 6, h = blockIdx.x & 63;
  const float* xb = x + (size_t)b * 262144 + (size_t)h * 64;   // [c][w] at c*4096+w
  for (int i = tid; i < 4096; i += 256) {                      // coalesced, transpose
    const int c = i >> 6, w = i & 63;
    xin[w * 65 + c] = xb[(size_t)c * 4096 + w];
  }
  __syncthreads();
  const float* fq = freq + (size_t)h * 4096;                   // [w][c]
  for (int i = tid; i < 4096; i += 256) {
    const int w = i >> 6, c = i & 63;
    xin[w * 65 + c] += fq[i];
    lB[w * LDB + c] = WkT[i];                                  // B^T = Wk^T [kd][c]
  }
  __syncthreads();
  if (tid < 64) {
    float s = 0.0f, q = 0.0f;
    for (int c = 0; c < 64; ++c) { const float v = xin[tid * 65 + c]; s += v; q += v * v; }
    const float m = s * (1.0f / 64.0f);
    mu[tid] = m;
    rstd[tid] = rsqrtf(q * (1.0f / 64.0f) - m * m + 1e-5f);
  }
  __syncthreads();
  for (int i = tid; i < 4096; i += 256) {
    const int w = i >> 6, c = i & 63;
    lA[w * LDB + c] = f2bf((xin[w * 65 + c] - mu[w]) * rstd[w] * g_k[c] + b_k[c]);
  }
  __syncthreads();
  v8f acc[2]; zero_acc(acc);
  wmma64(lA, lB, acc, tid >> 5, tid & 31);
  store_acc_bf16(acc, tid >> 5, tid & 31,
                 buf + (size_t)b * 262144 + (size_t)h * 4096, 64, bk);
}

// ---------------------------------------------------------------------------
// K2/K4: DCT along dim0 of buf viewed as [b][r][j][c] (cos is symmetric, so
// forward-H and inverse-N are identical).  In place: block (b,j) owns region
// buf[b][*][j][*]; all reads are staged to LDS before any store.
// ---------------------------------------------------------------------------
__global__ void k_dct0(const bf16* __restrict__ cosb, bf16* __restrict__ buf) {
  __shared__ __align__(16) bf16 lA[64 * LDB];
  __shared__ __align__(16) bf16 lB[64 * LDB];
  const int tid = threadIdx.x;
  const int b = blockIdx.x >> 6, j = blockIdx.x & 63;
  bf16* s = buf + (size_t)b * 262144 + (size_t)j * 64;   // [k][c] at k*4096+c
  for (int i = tid; i < 4096; i += 256) {
    const int r = i >> 6, c = i & 63;
    lA[r * LDB + c] = cosb[i];
    lB[c * LDB + r] = s[(size_t)r * 4096 + c];           // B^T [c][k]
  }
  __syncthreads();
  v8f acc[2]; zero_acc(acc);
  wmma64(lA, lB, acc, tid >> 5, tid & 31);
  store_acc_bf16(acc, tid >> 5, tid & 31, s, 4096, nullptr);
}

// ---------------------------------------------------------------------------
// K3: DCT along W + spectral gate.  Block (b,n): T2[m][c] = sum_w cos[m][w]*
// T1[n][w][c], then *= wexp[n][m][c].  In place on buf[b][n][*][*].
// ---------------------------------------------------------------------------
__global__ void k_dctw_gate(const bf16* __restrict__ cosb, bf16* __restrict__ buf,
                            const float* __restrict__ wexp) {
  __shared__ __align__(16) bf16 lA[64 * LDB];
  __shared__ __align__(16) bf16 lB[64 * LDB];
  const int tid = threadIdx.x;
  const int b = blockIdx.x >> 6, n = blockIdx.x & 63;
  bf16* s = buf + (size_t)b * 262144 + (size_t)n * 4096;   // [w][c] contiguous
  for (int i = tid; i < 4096; i += 256) {
    const int w = i >> 6, c = i & 63;
    lA[w * LDB + c] = cosb[i];
    lB[c * LDB + w] = s[i];                                 // B^T [c][w]
  }
  __syncthreads();
  v8f acc[2]; zero_acc(acc);
  const int wave = tid >> 5, lane = tid & 31;
  wmma64(lA, lB, acc, wave, lane);
  const float* wx = wexp + (size_t)n * 4096;
  const int m0 = (wave >> 1) * 16 + ((lane >> 4) << 3);
  const int nb = (wave & 1) << 1;
#pragma unroll
  for (int t = 0; t < 2; ++t) {
    const int c = (nb + t) * 16 + (lane & 15);
#pragma unroll
    for (int r = 0; r < 8; ++r) {
      const int m = m0 + r;
      s[m * 64 + c] = f2bf(acc[t][r] * wx[m * 64 + c]);
    }
  }
}

// ---------------------------------------------------------------------------
// K5: per (b,h): inverse DCT over W -> T4[w][c] (f32 in LDS), LN over c,
// * silu(z)[h][w][c], GEMM @ Wo, + bo, store transposed to out[b][c][h][w].
// ---------------------------------------------------------------------------
__global__ void k_out(const bf16* __restrict__ cosb, const bf16* __restrict__ buf,
                      const float* __restrict__ sz,
                      const float* __restrict__ g_o, const float* __restrict__ b_o,
                      const bf16* __restrict__ WoT, const float* __restrict__ bo,
                      float* __restrict__ out) {
  __shared__ __align__(16) bf16 lA[64 * LDB];
  __shared__ __align__(16) bf16 lB[64 * LDB];
  __shared__ float sT[64 * 65];
  __shared__ float mu[64], rstd[64];
  const int tid = threadIdx.x;
  const int b = blockIdx.x >> 6, h = blockIdx.x & 63;
  const bf16* s = buf + (size_t)b * 262144 + (size_t)h * 4096;  // T3 [m][c]
  for (int i = tid; i < 4096; i += 256) {
    const int m = i >> 6, c = i & 63;
    lA[m * LDB + c] = cosb[i];
    lB[c * LDB + m] = s[i];                                     // B^T [c][m]
  }
  __syncthreads();
  const int wave = tid >> 5, lane = tid & 31;
  const int m0 = (wave >> 1) * 16 + ((lane >> 4) << 3);
  const int nb = (wave & 1) << 1;
  v8f acc[2]; zero_acc(acc);
  wmma64(lA, lB, acc, wave, lane);
#pragma unroll
  for (int t = 0; t < 2; ++t) {
    const int n = (nb + t) * 16 + (lane & 15);
#pragma unroll
    for (int r = 0; r < 8; ++r) sT[(m0 + r) * 65 + n] = acc[t][r];  // T4[w][c]
  }
  __syncthreads();
  if (tid < 64) {
    float s2 = 0.0f, q = 0.0f;
    for (int c = 0; c < 64; ++c) { const float v = sT[tid * 65 + c]; s2 += v; q += v * v; }
    const float m = s2 * (1.0f / 64.0f);
    mu[tid] = m;
    rstd[tid] = rsqrtf(q * (1.0f / 64.0f) - m * m + 1e-5f);
  }
  __syncthreads();
  const float* szb = sz + (size_t)h * 4096;
  for (int i = tid; i < 4096; i += 256) {
    const int w = i >> 6, c = i & 63;
    const float g = (sT[w * 65 + c] - mu[w]) * rstd[w] * g_o[c] + b_o[c];
    lA[w * LDB + c] = f2bf(g * szb[i]);                        // A = gated [w][kd]
    lB[w * LDB + c] = WoT[i];                                  // B^T = Wo^T [cout][kd]
  }
  __syncthreads();
  v8f acc2[2]; zero_acc(acc2);
  wmma64(lA, lB, acc2, wave, lane);
#pragma unroll
  for (int t = 0; t < 2; ++t) {
    const int n = (nb + t) * 16 + (lane & 15);
#pragma unroll
    for (int r = 0; r < 8; ++r) sT[n * 65 + (m0 + r)] = acc2[t][r];  // O^T [cout][w]
  }
  __syncthreads();
  float* ob = out + (size_t)b * 262144 + (size_t)h * 64;       // out[b][c][h][w]
  for (int i = tid; i < 4096; i += 256) {
    const int c = i >> 6, w = i & 63;
    ob[(size_t)c * 4096 + w] = sT[c * 65 + w] + bo[c];         // coalesced in w
  }
}

// ---------------------------------------------------------------------------
extern "C" void kernel_launch(void* const* d_in, const int* in_sizes, int n_in,
                              void* d_out, int out_size, void* d_ws, size_t ws_size,
                              hipStream_t stream) {
  (void)in_sizes; (void)n_in; (void)out_size; (void)ws_size;
  const float* x    = (const float*)d_in[0];
  const float* freq = (const float*)d_in[1];
  const float* g_k  = (const float*)d_in[2];
  const float* b_k  = (const float*)d_in[3];
  const float* Wk   = (const float*)d_in[4];
  const float* bk   = (const float*)d_in[5];
  const float* g_z  = (const float*)d_in[6];
  const float* b_z  = (const float*)d_in[7];
  const float* Wz   = (const float*)d_in[8];
  const float* bz   = (const float*)d_in[9];
  const float* g_o  = (const float*)d_in[10];
  const float* b_o  = (const float*)d_in[11];
  const float* Wo   = (const float*)d_in[12];
  const float* bo   = (const float*)d_in[13];
  float* out = (float*)d_out;

  char* ws = (char*)d_ws;
  bf16*  buf  = (bf16*)(ws);                 // 33554432 B : k/T1/T2/T3 in place
  float* wexp = (float*)(ws + 33554432);     //  1048576 B
  float* sz   = (float*)(ws + 34603008);     //  1048576 B
  bf16*  cosb = (bf16*)(ws + 35651584);      //     8192 B
  bf16*  WkT  = (bf16*)(ws + 35659776);      //     8192 B
  bf16*  WoT  = (bf16*)(ws + 35667968);      //     8192 B

  k_tables   <<<64,   64,  0, stream>>>(Wk, Wo, cosb, WkT, WoT);
  k_zgate    <<<4096, 64,  0, stream>>>(freq, g_z, b_z, Wz, bz, wexp, sz);
  k_input    <<<4096, 256, 0, stream>>>(x, freq, g_k, b_k, bk, WkT, buf);
  k_dct0     <<<4096, 256, 0, stream>>>(cosb, buf);        // forward DCT over H
  k_dctw_gate<<<4096, 256, 0, stream>>>(cosb, buf, wexp);  // forward DCT over W + gate
  k_dct0     <<<4096, 256, 0, stream>>>(cosb, buf);        // inverse DCT over N
  k_out      <<<4096, 256, 0, stream>>>(cosb, buf, sz, g_o, b_o, WoT, bo, out);
}
